// VirtualNodeMPNNUpdate_37134287242014
// MI455X (gfx1250) — matrix operand order repaired and checked by reading, not compiled, not run on hardware
//
#include <hip/hip_runtime.h>
#include <stdint.h>

#define NN 50000
#define NB 50
#define NV 16
#define NC 128
#define NTILES (NN / 16) // 3125

typedef __attribute__((ext_vector_type(16))) __bf16 v16bf;
typedef __attribute__((ext_vector_type(16))) float  v16f;
typedef __attribute__((ext_vector_type(8)))  float  v8f;

union BfOp { v16bf v; uint32_t u[8]; };

// Native RNE f32 -> bf16 via hardware convert (v_cvt_pk_bf16_f32).
__device__ __forceinline__ void packB(BfOp& bo, const float* vv) {
  v16f f;
#pragma unroll
  for (int i = 0; i < 16; ++i) f[i] = vv[i];
  bo.v = __builtin_convertvector(f, v16bf);
}

// Load one weight A-operand tile (t, kk) from row-major bf16 weights.
__device__ __forceinline__ void load_wtile(BfOp& ao, const uint16_t* __restrict__ W,
                                           int t, int kk, int lane) {
  const int r  = lane & 15;
  const int h8 = (lane & 16) ? 16 : 0; // k order used here for A (B matches): lanes<16 hold K 0..15, lanes>=16 hold K 16..31
  const uint16_t* p = W + ((t * 16 + r) * NC + kk * 32 + h8);
  uint4 c1 = *(const uint4*)p;
  uint4 c2 = *(const uint4*)(p + 8);
  ao.u[0] = c1.x; ao.u[1] = c1.y; ao.u[2] = c1.z; ao.u[3] = c1.w;
  ao.u[4] = c2.x; ao.u[5] = c2.y; ao.u[6] = c2.z; ao.u[7] = c2.w;
}

// Preload a full 128x128 bf16 weight matrix into 256 VGPRs (loop-invariant).
__device__ __forceinline__ void load_wreg(BfOp (&wr)[4][8], const uint16_t* __restrict__ W,
                                          int lane) {
#pragma unroll
  for (int kk = 0; kk < 4; ++kk)
#pragma unroll
    for (int t = 0; t < 8; ++t) load_wtile(wr[kk][t], W, t, kk, lane);
}

// One K=32 step with register-resident weights: 8 independent back-to-back WMMA.
__device__ __forceinline__ void gemm_step_r(v8f* acc, const BfOp (&wr)[4][8],
                                            const BfOp& bop, int kk) {
#pragma unroll
  for (int t = 0; t < 8; ++t)
    acc[t] = __builtin_amdgcn_wmma_f32_16x16x32_bf16(false, wr[kk][t].v, false, bop.v,
                                                     (short)0, acc[t], false, false);
}

// One K=32 step loading weights from global (single-shot kernels).
__device__ __forceinline__ void gemm_step(v8f* acc, const uint16_t* __restrict__ W,
                                          const BfOp& bop, int kk, int lane) {
#pragma unroll
  for (int t = 0; t < 8; ++t) {
    BfOp ao;
    load_wtile(ao, W, t, kk, lane);
    acc[t] = __builtin_amdgcn_wmma_f32_16x16x32_bf16(false, ao.v, false, bop.v,
                                                     (short)0, acc[t], false, false);
  }
}

// Convert C-layout accumulators to B-operand-layout f32 for k-step kk.
// Lane pair (l, l^16) holds complementary halves -> two shfl_xor(16) per reg.
__device__ __forceinline__ void convC2B(const v8f* acc, int kk, bool lo,
                                        const float* __restrict__ bias16,
                                        float* vv, bool relu) {
#pragma unroll
  for (int j = 0; j < 8; ++j) {
    float a  = acc[2 * kk][j];
    float bq = acc[2 * kk + 1][j];
    float pA = __shfl_xor(a, 16);
    float pB = __shfl_xor(bq, 16);
    vv[j]     = lo ? a  : pB;
    vv[8 + j] = lo ? pA : bq;
  }
#pragma unroll
  for (int i = 0; i < 4; ++i) {
    float4 b4 = ((const float4*)bias16)[i];
    vv[4 * i + 0] += b4.x; vv[4 * i + 1] += b4.y;
    vv[4 * i + 2] += b4.z; vv[4 * i + 3] += b4.w;
  }
  if (relu) {
#pragma unroll
    for (int i = 0; i < 16; ++i) vv[i] = fmaxf(vv[i], 0.f);
  }
}

// ---------------- small utility kernels ----------------
__global__ void k_zero(float* p, int n) {
  int i = blockIdx.x * blockDim.x + threadIdx.x;
  if (i < n) p[i] = 0.f;
}
__global__ void k_cnt(const int* __restrict__ batch, const int* __restrict__ mask,
                      float* __restrict__ cnt) {
  int i = blockIdx.x * blockDim.x + threadIdx.x;
  if (i < NN && mask[i]) atomicAdd(&cnt[batch[i]], 1.f);
}
// f32 -> bf16 weight conversion, 128x128 dest, src row stride = ld (256 for w1 halves)
__global__ void k_wconv(const float* __restrict__ src, int ld, uint16_t* __restrict__ dst) {
  int i = blockIdx.x * blockDim.x + threadIdx.x;
  if (i < NC * NC) {
    int r = i >> 7, c = i & 127;
    union { __bf16 h; uint16_t u; } cv;
    cv.h = (__bf16)src[r * ld + c];
    dst[i] = cv.u;
  }
}
// out[r][o] = sum_k in[r*NC+k] * W[o*wld + woff + k] + bias[o]   (800 rows)
__global__ void __launch_bounds__(128) k_rowlin(const float* __restrict__ in,
                                                const float* __restrict__ W, int wld, int woff,
                                                const float* __restrict__ bias,
                                                float* __restrict__ out) {
  int r = blockIdx.x, o = threadIdx.x;
  float acc = bias[o];
  const float* ip = in + r * NC;
  const float* wp = W + o * wld + woff;
  for (int k = 0; k < NC; ++k) acc = fmaf(ip[k], wp[k], acc);
  out[r * NC + o] = acc;
}

// ---------------- WMMA precompute: A_g = nodef @ W1gaT, A_s = nodef @ W1saT ----
__global__ void __launch_bounds__(128) k_pre(const float* __restrict__ nodef,
                                             const uint16_t* __restrict__ Wg,
                                             const uint16_t* __restrict__ Ws,
                                             float* __restrict__ Ag, float* __restrict__ As) {
  const int lane = threadIdx.x & 31;
  const int tile = blockIdx.x * 4 + (threadIdx.x >> 5);
  if (tile >= NTILES) return;
  const int m = lane & 15;
  const bool lo = lane < 16;
  const int s16 = lo ? 0 : 16;
  const int n = tile * 16 + m;

  v8f ag[8], as_[8];
#pragma unroll
  for (int t = 0; t < 8; ++t)
#pragma unroll
    for (int j = 0; j < 8; ++j) { ag[t][j] = 0.f; as_[t][j] = 0.f; }

#pragma unroll
  for (int kk = 0; kk < 4; ++kk) {
    float vv[16];
    const float4* p = (const float4*)(nodef + n * NC + kk * 32 + s16);
#pragma unroll
    for (int i = 0; i < 4; ++i) {
      float4 x = p[i];
      vv[4 * i + 0] = x.x; vv[4 * i + 1] = x.y; vv[4 * i + 2] = x.z; vv[4 * i + 3] = x.w;
    }
    BfOp bo; packB(bo, vv);
    gemm_step(ag, Wg, bo, kk, lane);
    gemm_step(as_, Ws, bo, kk, lane);
  }
  const int c8 = lo ? 0 : 8;
#pragma unroll
  for (int t = 0; t < 8; ++t)
#pragma unroll
    for (int j = 0; j < 8; ++j) {
      int c = t * 16 + j + c8;
      Ag[n * NC + c] = ag[t][j];
      As[n * NC + c] = as_[t][j];
    }
}

// ---------------- gather MPNN main kernel (weights register-resident) --------
__global__ void __launch_bounds__(128) k_gather(const float* __restrict__ Ag,
                                                const float* __restrict__ Bvg,
                                                const int* __restrict__ batch,
                                                const int* __restrict__ mask,
                                                const uint16_t* __restrict__ W2,
                                                const uint16_t* __restrict__ W3,
                                                const float* __restrict__ b2,
                                                float* __restrict__ sums) {
  const int lane = threadIdx.x & 31;
  const int wave = blockIdx.x * 4 + (threadIdx.x >> 5);
  const int nw = gridDim.x * 4;
  const int per = (NN + nw - 1) / nw;
  const int n0 = wave * per;
  const int n1 = (n0 + per < NN) ? (n0 + per) : NN;
  const int m = lane & 15;
  const bool lo = lane < 16;
  const int s16 = lo ? 0 : 16;
  const int c8 = lo ? 0 : 8;

  BfOp w2r[4][8], w3r[4][8]; // 512 VGPRs of loop-invariant weights
  load_wreg(w2r, W2, lane);
  load_wreg(w3r, W3, lane);

  v8f accb[8];
#pragma unroll
  for (int t = 0; t < 8; ++t)
#pragma unroll
    for (int j = 0; j < 8; ++j) accb[t][j] = 0.f;
  int curb = -1;

  auto flush = [&](int bb) {
#pragma unroll
    for (int t = 0; t < 8; ++t)
#pragma unroll
      for (int j = 0; j < 8; ++j) {
        int c = t * 16 + j + c8;
        atomicAdd(&sums[(bb * NV + m) * NC + c], accb[t][j]);
        accb[t][j] = 0.f;
      }
  };

  for (int n = n0; n < n1; ++n) {
    int b = batch[n];
    if (b != curb) { if (curb >= 0) flush(curb); curb = b; }
    if (!mask[n]) continue; // mask multiplies msg; mask in {0,1} -> skip
    // GEMM1: h2pre^T = W2 @ h1^T, h1 = relu(Ag[n] + Bvg[b,v]) built in-register
    v8f a1[8];
#pragma unroll
    for (int t = 0; t < 8; ++t)
#pragma unroll
      for (int j = 0; j < 8; ++j) a1[t][j] = 0.f;
#pragma unroll
    for (int kk = 0; kk < 4; ++kk) {
      float vv[16];
      const float4* ap = (const float4*)(Ag + n * NC + kk * 32 + s16);
      const float4* bp = (const float4*)(Bvg + (b * NV + m) * NC + kk * 32 + s16);
#pragma unroll
      for (int i = 0; i < 4; ++i) {
        float4 x = ap[i], y = bp[i];
        vv[4 * i + 0] = fmaxf(x.x + y.x, 0.f);
        vv[4 * i + 1] = fmaxf(x.y + y.y, 0.f);
        vv[4 * i + 2] = fmaxf(x.z + y.z, 0.f);
        vv[4 * i + 3] = fmaxf(x.w + y.w, 0.f);
      }
      BfOp bo; packB(bo, vv);
      gemm_step_r(a1, w2r, bo, kk);
    }
    // GEMM2: msgpre^T = W3 @ relu(a1 + b2)^T  (b3 deferred to vn kernel)
    v8f a2[8];
#pragma unroll
    for (int t = 0; t < 8; ++t)
#pragma unroll
      for (int j = 0; j < 8; ++j) a2[t][j] = 0.f;
#pragma unroll
    for (int kk = 0; kk < 4; ++kk) {
      float vv[16];
      convC2B(a1, kk, lo, b2 + kk * 32 + s16, vv, true);
      BfOp bo; packB(bo, vv);
      gemm_step_r(a2, w3r, bo, kk);
    }
#pragma unroll
    for (int t = 0; t < 8; ++t)
#pragma unroll
      for (int j = 0; j < 8; ++j) accb[t][j] += a2[t][j];
  }
  if (curb >= 0) flush(curb);
}

// ---------------- scatter MPNN main kernel (weights register-resident) -------
__global__ void __launch_bounds__(128) k_scatter(const float* __restrict__ As,
                                                 const float* __restrict__ Bvs,
                                                 const int* __restrict__ batch,
                                                 const int* __restrict__ mask,
                                                 const uint16_t* __restrict__ W2,
                                                 const uint16_t* __restrict__ W3,
                                                 const float* __restrict__ b2,
                                                 const float* __restrict__ b3,
                                                 float* __restrict__ upd) {
  const int lane = threadIdx.x & 31;
  const int wave = blockIdx.x * 4 + (threadIdx.x >> 5);
  const int nw = gridDim.x * 4;
  const int per = (NN + nw - 1) / nw;
  const int n0 = wave * per;
  const int n1 = (n0 + per < NN) ? (n0 + per) : NN;
  const int m = lane & 15;
  const bool lo = lane < 16;
  const int s16 = lo ? 0 : 16;
  const int c8 = lo ? 0 : 8;

  BfOp w2r[4][8], w3r[4][8];
  load_wreg(w2r, W2, lane);
  load_wreg(w3r, W3, lane);

  for (int n = n0; n < n1; ++n) {
    if (!mask[n]) { // node_update = 0
      *(float4*)(upd + n * NC + lane * 4) = make_float4(0.f, 0.f, 0.f, 0.f);
      continue;
    }
    int b = batch[n];
    v8f a1[8], a2[8];
#pragma unroll
    for (int t = 0; t < 8; ++t)
#pragma unroll
      for (int j = 0; j < 8; ++j) { a1[t][j] = 0.f; a2[t][j] = 0.f; }
#pragma unroll
    for (int kk = 0; kk < 4; ++kk) {
      float vv[16];
      const float4* ap = (const float4*)(As + n * NC + kk * 32 + s16);
      const float4* bp = (const float4*)(Bvs + (b * NV + m) * NC + kk * 32 + s16);
#pragma unroll
      for (int i = 0; i < 4; ++i) {
        float4 x = ap[i], y = bp[i];
        vv[4 * i + 0] = fmaxf(x.x + y.x, 0.f);
        vv[4 * i + 1] = fmaxf(x.y + y.y, 0.f);
        vv[4 * i + 2] = fmaxf(x.z + y.z, 0.f);
        vv[4 * i + 3] = fmaxf(x.w + y.w, 0.f);
      }
      BfOp bo; packB(bo, vv);
      gemm_step_r(a1, w2r, bo, kk);
    }
#pragma unroll
    for (int kk = 0; kk < 4; ++kk) {
      float vv[16];
      convC2B(a1, kk, lo, b2 + kk * 32 + s16, vv, true);
      BfOp bo; packB(bo, vv);
      gemm_step_r(a2, w3r, bo, kk);
    }
    // mean over v (columns): reduce within each 16-lane half
#pragma unroll
    for (int t = 0; t < 8; ++t)
#pragma unroll
      for (int j = 0; j < 8; ++j) {
        float s = a2[t][j];
        s += __shfl_xor(s, 1); s += __shfl_xor(s, 2);
        s += __shfl_xor(s, 4); s += __shfl_xor(s, 8);
        if (m == 0) {
          int c = t * 16 + j + c8;
          upd[n * NC + c] = s * (1.f / 16.f) + b3[c];
        }
      }
  }
}

// ---------------- node finalize: LN1 -> MLP3 (WMMA) -> residual -> LN2 -------
__global__ void __launch_bounds__(128) k_nodefin(const float* __restrict__ nodef,
                                                 const float* __restrict__ upd,
                                                 const uint16_t* __restrict__ W1,
                                                 const uint16_t* __restrict__ W2,
                                                 const uint16_t* __restrict__ W3,
                                                 const float* __restrict__ b1,
                                                 const float* __restrict__ b2,
                                                 const float* __restrict__ b3,
                                                 const float* __restrict__ g1,
                                                 const float* __restrict__ be1,
                                                 const float* __restrict__ g2,
                                                 const float* __restrict__ be2,
                                                 float* __restrict__ out) {
  const int lane = threadIdx.x & 31;
  const int tile = blockIdx.x * 4 + (threadIdx.x >> 5);
  if (tile >= NTILES) return;
  const int m = lane & 15;
  const bool lo = lane < 16;
  const int s16 = lo ? 0 : 16;
  const int n = tile * 16 + m;

  float xr[4][16];
  float s = 0.f, ss = 0.f;
#pragma unroll
  for (int kk = 0; kk < 4; ++kk) {
    const float4* p = (const float4*)(nodef + n * NC + kk * 32 + s16);
    const float4* q = (const float4*)(upd + n * NC + kk * 32 + s16);
#pragma unroll
    for (int i = 0; i < 4; ++i) {
      float4 a = p[i], bq = q[i];
      float e0 = a.x + bq.x, e1 = a.y + bq.y, e2 = a.z + bq.z, e3 = a.w + bq.w;
      xr[kk][4 * i + 0] = e0; xr[kk][4 * i + 1] = e1;
      xr[kk][4 * i + 2] = e2; xr[kk][4 * i + 3] = e3;
      s += e0 + e1 + e2 + e3;
      ss += e0 * e0 + e1 * e1 + e2 * e2 + e3 * e3;
    }
  }
  s += __shfl_xor(s, 16); ss += __shfl_xor(ss, 16); // lane pair covers full row
  float mean = s * (1.f / NC);
  float rstd = rsqrtf(ss * (1.f / NC) - mean * mean + 1e-5f);
#pragma unroll
  for (int kk = 0; kk < 4; ++kk) {
    const float4* gp = (const float4*)(g1 + kk * 32 + s16);
    const float4* bp = (const float4*)(be1 + kk * 32 + s16);
#pragma unroll
    for (int i = 0; i < 4; ++i) {
      float4 g4 = gp[i], b4 = bp[i];
      xr[kk][4 * i + 0] = (xr[kk][4 * i + 0] - mean) * rstd * g4.x + b4.x;
      xr[kk][4 * i + 1] = (xr[kk][4 * i + 1] - mean) * rstd * g4.y + b4.y;
      xr[kk][4 * i + 2] = (xr[kk][4 * i + 2] - mean) * rstd * g4.z + b4.z;
      xr[kk][4 * i + 3] = (xr[kk][4 * i + 3] - mean) * rstd * g4.w + b4.w;
    }
  }
  v8f a1[8], a2[8], a3[8];
#pragma unroll
  for (int t = 0; t < 8; ++t)
#pragma unroll
    for (int j = 0; j < 8; ++j) { a1[t][j] = 0.f; a2[t][j] = 0.f; a3[t][j] = 0.f; }
#pragma unroll
  for (int kk = 0; kk < 4; ++kk) { BfOp bo; packB(bo, xr[kk]); gemm_step(a1, W1, bo, kk, lane); }
#pragma unroll
  for (int kk = 0; kk < 4; ++kk) {
    float vv[16]; convC2B(a1, kk, lo, b1 + kk * 32 + s16, vv, true);
    BfOp bo; packB(bo, vv); gemm_step(a2, W2, bo, kk, lane);
  }
#pragma unroll
  for (int kk = 0; kk < 4; ++kk) {
    float vv[16]; convC2B(a2, kk, lo, b2 + kk * 32 + s16, vv, true);
    BfOp bo; packB(bo, vv); gemm_step(a3, W3, bo, kk, lane);
  }
  // residual + LN2
  float yr[4][16];
  s = 0.f; ss = 0.f;
#pragma unroll
  for (int kk = 0; kk < 4; ++kk) {
    convC2B(a3, kk, lo, b3 + kk * 32 + s16, yr[kk], false);
#pragma unroll
    for (int i = 0; i < 16; ++i) {
      float y = yr[kk][i] + xr[kk][i];
      yr[kk][i] = y;
      s += y; ss += y * y;
    }
  }
  s += __shfl_xor(s, 16); ss += __shfl_xor(ss, 16);
  mean = s * (1.f / NC);
  rstd = rsqrtf(ss * (1.f / NC) - mean * mean + 1e-5f);
#pragma unroll
  for (int kk = 0; kk < 4; ++kk) {
    float4* op = (float4*)(out + n * NC + kk * 32 + s16);
    const float4* gp = (const float4*)(g2 + kk * 32 + s16);
    const float4* bp = (const float4*)(be2 + kk * 32 + s16);
#pragma unroll
    for (int i = 0; i < 4; ++i) {
      float4 g4 = gp[i], b4 = bp[i], o4;
      o4.x = (yr[kk][4 * i + 0] - mean) * rstd * g4.x + b4.x;
      o4.y = (yr[kk][4 * i + 1] - mean) * rstd * g4.y + b4.y;
      o4.z = (yr[kk][4 * i + 2] - mean) * rstd * g4.z + b4.z;
      o4.w = (yr[kk][4 * i + 3] - mean) * rstd * g4.w + b4.w;
      op[i] = o4;
    }
  }
}

// ---------------- vn pipeline (tiny: [50,16,128]) ----------------
struct TL {
  const float *in_w, *in_b, *out_w, *out_b, *f1w, *f1b, *f2w, *f2b, *l1g, *l1b, *l2g, *l2b;
};
struct VnA {
  const float *vnf, *sums, *cnt, *gb3;
  const float *vl1g, *vl1b, *vl2g, *vl2b;
  const float *fw1, *fb1, *fw2, *fb2, *fw3, *fb3;
  TL L0, L1;
  float* out;
};

__device__ void d_lin(const float (*src)[NC], float (*dst)[NC], const float* W,
                      const float* bias, bool relu, int th) {
  for (int v = 0; v < NV; ++v) {
    float acc = bias[th];
    for (int k = 0; k < NC; ++k) acc = fmaf(src[v][k], W[th * NC + k], acc);
    dst[v][th] = relu ? fmaxf(acc, 0.f) : acc;
  }
  __syncthreads();
}
__device__ void d_ln(float (*x)[NC], const float* g, const float* bt, int th) {
  __syncthreads();
  if (th < NV) {
    float s = 0.f;
    for (int c = 0; c < NC; ++c) s += x[th][c];
    float mean = s * (1.f / NC), ss = 0.f;
    for (int c = 0; c < NC; ++c) { float d = x[th][c] - mean; ss += d * d; }
    float rstd = rsqrtf(ss * (1.f / NC) + 1e-5f);
    for (int c = 0; c < NC; ++c) x[th][c] = (x[th][c] - mean) * rstd * g[c] + bt[c];
  }
  __syncthreads();
}
__device__ void d_encoder(float (*x)[NC], float (*qb)[NC], float (*kb)[NC],
                          float (*vb)[NC], float (*sc)[NV][NV], const TL& L, int th) {
  for (int part = 0; part < 3; ++part) {
    int o = part * NC + th;
    for (int v = 0; v < NV; ++v) {
      float acc = L.in_b[o];
      for (int k = 0; k < NC; ++k) acc = fmaf(x[v][k], L.in_w[o * NC + k], acc);
      (part == 0 ? qb : part == 1 ? kb : vb)[v][th] = acc;
    }
  }
  __syncthreads();
  for (int i = 0; i < 8; ++i) { // 4h x 16q x 16k scores
    int id = th + i * 128;
    int h = id >> 8, q = (id >> 4) & 15, kx = id & 15;
    float acc = 0.f;
    for (int d = 0; d < 32; ++d) acc = fmaf(qb[q][h * 32 + d], kb[kx][h * 32 + d], acc);
    sc[h][q][kx] = acc * 0.17677669529663687f; // 1/sqrt(32)
  }
  __syncthreads();
  if (th < 64) { // softmax rows
    int h = th >> 4, q = th & 15;
    float mx = -1e30f;
    for (int kx = 0; kx < 16; ++kx) mx = fmaxf(mx, sc[h][q][kx]);
    float sm = 0.f;
    for (int kx = 0; kx < 16; ++kx) { float e = __expf(sc[h][q][kx] - mx); sc[h][q][kx] = e; sm += e; }
    float inv = 1.f / sm;
    for (int kx = 0; kx < 16; ++kx) sc[h][q][kx] *= inv;
  }
  __syncthreads();
  { // o = att @ v  -> into qb
    int h = th >> 5;
    for (int q = 0; q < NV; ++q) {
      float acc = 0.f;
      for (int kx = 0; kx < 16; ++kx) acc = fmaf(sc[h][q][kx], vb[kx][th], acc);
      qb[q][th] = acc;
    }
  }
  __syncthreads();
  d_lin(qb, kb, L.out_w, L.out_b, false, th);
  for (int v = 0; v < NV; ++v) x[v][th] += kb[v][th];
  d_ln(x, L.l1g, L.l1b, th);
  d_lin(x, qb, L.f1w, L.f1b, true, th);
  d_lin(qb, kb, L.f2w, L.f2b, false, th);
  for (int v = 0; v < NV; ++v) x[v][th] += kb[v][th];
  d_ln(x, L.l2g, L.l2b, th);
}

__global__ void __launch_bounds__(128) k_vn(VnA A) {
  __shared__ float xb[NV][NC], sq[NV][NC], sk[NV][NC], sv[NV][NC];
  __shared__ float sc[4][NV][NV];
  int th = threadIdx.x, b = blockIdx.x;
  float cb = A.cnt[b];
  float inv = 1.f / fmaxf(cb, 1.f);
  float has = cb > 0.5f ? 1.f : 0.f;
  for (int v = 0; v < NV; ++v) { // vn_update = sums/clip(cnt,1) + b3*(cnt>0)
    int idx = (b * NV + v) * NC + th;
    xb[v][th] = A.vnf[idx] + A.sums[idx] * inv + has * A.gb3[th];
  }
  d_ln(xb, A.vl1g, A.vl1b, th);
  d_lin(xb, sq, A.fw1, A.fb1, true, th);
  d_lin(sq, sk, A.fw2, A.fb2, true, th);
  d_lin(sk, sv, A.fw3, A.fb3, false, th);
  for (int v = 0; v < NV; ++v) xb[v][th] += sv[v][th];
  d_ln(xb, A.vl2g, A.vl2b, th);
  d_encoder(xb, sq, sk, sv, sc, A.L0, th);
  d_encoder(xb, sq, sk, sv, sc, A.L1, th);
  for (int v = 0; v < NV; ++v) A.out[(b * NV + v) * NC + th] = xb[v][th];
}

// ---------------- host launcher ----------------
extern "C" void kernel_launch(void* const* d_in, const int* in_sizes, int n_in,
                              void* d_out, int out_size, void* d_ws, size_t ws_size,
                              hipStream_t stream) {
  (void)in_sizes; (void)n_in; (void)out_size; (void)ws_size;
  const float* nodef = (const float*)d_in[0];
  const float* vnf   = (const float*)d_in[1];
  const int*   batch = (const int*)d_in[2];
  const int*   mask  = (const int*)d_in[3];
  const float *gw1 = (const float*)d_in[4],  *gb1 = (const float*)d_in[5];
  const float *gw2 = (const float*)d_in[6],  *gb2 = (const float*)d_in[7];
  const float *gw3 = (const float*)d_in[8],  *gb3 = (const float*)d_in[9];
  const float *sw1 = (const float*)d_in[10], *sb1 = (const float*)d_in[11];
  const float *sw2 = (const float*)d_in[12], *sb2 = (const float*)d_in[13];
  const float *sw3 = (const float*)d_in[14], *sb3 = (const float*)d_in[15];
  const float *vw1 = (const float*)d_in[16], *vb1 = (const float*)d_in[17];
  const float *vw2 = (const float*)d_in[18], *vb2 = (const float*)d_in[19];
  const float *vw3 = (const float*)d_in[20], *vb3 = (const float*)d_in[21];
  const float *nw1 = (const float*)d_in[22], *nb1 = (const float*)d_in[23];
  const float *nw2 = (const float*)d_in[24], *nb2 = (const float*)d_in[25];
  const float *nw3 = (const float*)d_in[26], *nb3 = (const float*)d_in[27];
  const float* tl0[12]; const float* tl1[12];
  for (int i = 0; i < 12; ++i) { tl0[i] = (const float*)d_in[28 + i]; tl1[i] = (const float*)d_in[40 + i]; }
  const float *vl1g = (const float*)d_in[52], *vl1b = (const float*)d_in[53];
  const float *vl2g = (const float*)d_in[54], *vl2b = (const float*)d_in[55];
  const float *nl1g = (const float*)d_in[56], *nl1b = (const float*)d_in[57];
  const float *nl2g = (const float*)d_in[58], *nl2b = (const float*)d_in[59];

  char* w = (char*)d_ws;
  auto alloc = [&](size_t bytes) { void* p = (void*)w; w += (bytes + 255) & ~(size_t)255; return p; };
  float* sums = (float*)alloc((size_t)NB * NV * NC * 4);
  float* cnt  = (float*)alloc(64 * 4);
  float* Ag   = (float*)alloc((size_t)NN * NC * 4);
  float* As   = (float*)alloc((size_t)NN * NC * 4);
  float* Bvg  = (float*)alloc((size_t)NB * NV * NC * 4);
  float* Bvs  = (float*)alloc((size_t)NB * NV * NC * 4);
  float* upd  = (float*)alloc((size_t)NN * NC * 4);
  uint16_t* Wga = (uint16_t*)alloc(NC * NC * 2);
  uint16_t* Wsa = (uint16_t*)alloc(NC * NC * 2);
  uint16_t* W2g = (uint16_t*)alloc(NC * NC * 2);
  uint16_t* W3g = (uint16_t*)alloc(NC * NC * 2);
  uint16_t* W2s = (uint16_t*)alloc(NC * NC * 2);
  uint16_t* W3s = (uint16_t*)alloc(NC * NC * 2);
  uint16_t* Wn1 = (uint16_t*)alloc(NC * NC * 2);
  uint16_t* Wn2 = (uint16_t*)alloc(NC * NC * 2);
  uint16_t* Wn3 = (uint16_t*)alloc(NC * NC * 2);

  float* out_nf = (float*)d_out;
  float* out_vn = out_nf + (size_t)NN * NC;

  k_zero<<<(NB * NV * NC + 64 + 255) / 256, 256, 0, stream>>>(sums, NB * NV * NC + 64);
  k_cnt<<<(NN + 255) / 256, 256, 0, stream>>>(batch, mask, cnt);
  k_wconv<<<64, 256, 0, stream>>>(gw1, 256, Wga); // first half of gather w1
  k_wconv<<<64, 256, 0, stream>>>(sw1, 256, Wsa); // first half of scatter w1
  k_wconv<<<64, 256, 0, stream>>>(gw2, 128, W2g);
  k_wconv<<<64, 256, 0, stream>>>(gw3, 128, W3g);
  k_wconv<<<64, 256, 0, stream>>>(sw2, 128, W2s);
  k_wconv<<<64, 256, 0, stream>>>(sw3, 128, W3s);
  k_wconv<<<64, 256, 0, stream>>>(nw1, 128, Wn1);
  k_wconv<<<64, 256, 0, stream>>>(nw2, 128, Wn2);
  k_wconv<<<64, 256, 0, stream>>>(nw3, 128, Wn3);
  // Bv_g = vn_features @ W1g[:,128:]^T + b1
  k_rowlin<<<NB * NV, 128, 0, stream>>>(vnf, gw1, 256, 128, gb1, Bvg);
  k_pre<<<(NTILES + 3) / 4, 128, 0, stream>>>(nodef, Wga, Wsa, Ag, As);
  k_gather<<<128, 128, 0, stream>>>(Ag, Bvg, batch, mask, W2g, W3g, gb2, sums);

  VnA A;
  A.vnf = vnf; A.sums = sums; A.cnt = cnt; A.gb3 = gb3;
  A.vl1g = vl1g; A.vl1b = vl1b; A.vl2g = vl2g; A.vl2b = vl2b;
  A.fw1 = vw1; A.fb1 = vb1; A.fw2 = vw2; A.fb2 = vb2; A.fw3 = vw3; A.fb3 = vb3;
  A.L0 = { tl0[0], tl0[1], tl0[2], tl0[3], tl0[4], tl0[5], tl0[6], tl0[7], tl0[8], tl0[9], tl0[10], tl0[11] };
  A.L1 = { tl1[0], tl1[1], tl1[2], tl1[3], tl1[4], tl1[5], tl1[6], tl1[7], tl1[8], tl1[9], tl1[10], tl1[11] };
  A.out = out_vn;
  k_vn<<<NB, 128, 0, stream>>>(A);

  // Bv_s = vn_refined @ W1s[:,128:]^T + b1  (reads refined vn from d_out)
  k_rowlin<<<NB * NV, 128, 0, stream>>>(out_vn, sw1, 256, 128, sb1, Bvs);
  k_scatter<<<128, 128, 0, stream>>>(As, Bvs, batch, mask, W2s, W3s, sb2, sb3, upd);
  k_nodefin<<<(NTILES + 3) / 4, 128, 0, stream>>>(nodef, upd, Wn1, Wn2, Wn3,
                                                  nb1, nb2, nb3, nl1g, nl1b, nl2g, nl2b, out_nf);
}
